// CluttrEncoder_53326313947351
// MI455X (gfx1250) — matrix operand order repaired and checked by reading, not compiled, not run on hardware
//
#include <hip/hip_runtime.h>

typedef __bf16 bf16_t;
typedef __bf16 v16bf __attribute__((ext_vector_type(16)));
typedef __bf16 v8bf  __attribute__((ext_vector_type(8)));
typedef float  v8f   __attribute__((ext_vector_type(8)));

// Problem constants
#define BATCH   128
#define SEQ     512
#define DIM     300
#define DPAD    320            // K padded to 10 k-tiles of 32
#define GDIM    1200           // 4*DIM fused gates
#define MROWS   (BATCH*SEQ)    // 65536
#define MTILES  (MROWS/16)     // 4096
#define KT_D    10             // k-tiles for K=300->320
#define NT_D    19             // live n-tiles for N=300
#define NT_DP   20             // padded to NB multiple (=DPAD/16)
#define NT_G    75             // live n-tiles for N=1200
#define NT_GP   76             // padded to NB multiple
#define NB      4              // N register blocking (n-tiles per wave)
#define LATENT  64

__device__ __forceinline__ float sigf(float x) { return 1.f / (1.f + __expf(-x)); }

// ---------------------------------------------------------------------------
// Pack an f32 row-major [Ksrc x Nsrc] weight into WMMA bf16 B-fragment tiles.
// Tile (kt,nt) = 32x16 (KxN). Per tile: 32 lanes x 16 bf16 contiguous, where
// lane l element e = W[kt*32 + (e<8 ? (l>>4)*8 + e : 16 + (l>>4)*8 + e-8)][nt*16 + (l&15)]  (zero-padded).
// Tiles ordered kt-fastest within an nt column: tile index = nt*KT + kt.
// Grid covers the PADDED n-tile count; pad tiles are all-zero.
// ---------------------------------------------------------------------------
__global__ void k_pack_w(const float* __restrict__ W, bf16_t* __restrict__ P,
                         int Ksrc, int Nsrc, int KT) {
    int kt = blockIdx.x, nt = blockIdx.y;
    int lane = threadIdx.x;                 // 0..31
    int hg = lane >> 4, n = nt * 16 + (lane & 15);
    bf16_t* dst = P + (((size_t)nt * KT + kt) << 9) + lane * 16;
#pragma unroll
    for (int e = 0; e < 16; ++e) {
        int k = kt * 32 + (e < 8 ? hg * 8 + e : 16 + hg * 8 + (e - 8));
        float v = (k < Ksrc && n < Nsrc) ? W[(size_t)k * Nsrc + n] : 0.f;
        dst[e] = (bf16_t)v;
    }
}

// ---------------------------------------------------------------------------
// Embedding gather -> bf16 activations, [M][DPAD], pad columns zeroed.
// ---------------------------------------------------------------------------
__global__ void k_embed(const int* __restrict__ tokens,
                        const float* __restrict__ embed,
                        bf16_t* __restrict__ X) {
    int i = blockIdx.x * 256 + threadIdx.x;      // over M*DPAD
    if (i >= MROWS * DPAD) return;
    int m = i / DPAD, d = i - m * DPAD;
    float v = 0.f;
    if (d < DIM) v = embed[(size_t)tokens[m] * DIM + d];
    X[i] = (bf16_t)v;
}

// ---------------------------------------------------------------------------
// WMMA A-fragment loader: row-major bf16 A, 16x32 bf16 fragment.
// lane (l&15) = row; elems 0..7 = K[hg*8 .. +7], elems 8..15 = K[16+hg*8 .. +7]
// ---------------------------------------------------------------------------
__device__ __forceinline__ v16bf load_a_frag(const bf16_t* __restrict__ ap) {
    v8bf lo = *(const v8bf*)ap;
    v8bf hi = *(const v8bf*)(ap + 16);
    v16bf a;
#pragma unroll
    for (int i = 0; i < 8; ++i) { a[i] = lo[i]; a[8 + i] = hi[i]; }
    return a;
}

// ---------------------------------------------------------------------------
// WMMA GEMM with N register blocking: out = act(A @ Bpacked + bias).
// block = 256 threads = 8 waves; each wave owns 1 M-tile x NB N-tiles:
// per k-tile it loads ONE A fragment and streams NB B fragments -> NB WMMAs.
// grid = (ceil(Mtiles/8), NTpad/NB). ACT: 0=none 1=relu 2=sigmoid.
// OUT_BF: store bf16 else f32. REMAP: row m=(b*S+s) -> row (s*Bn + b) (xg).
// Stores guarded by gn < Nout (padded N-tiles must not spill past the row).
// ---------------------------------------------------------------------------
template <int ACT, bool OUT_BF, bool REMAP>
__global__ void k_gemm(const bf16_t* __restrict__ Abase, int aRowStride,
                       const bf16_t* __restrict__ Bp, int KT,
                       const float* __restrict__ bias, int Nbias,
                       void* __restrict__ Out, int outRowStride, int Nout,
                       int Mtiles, int Ssz, int Bn) {
    int wave = threadIdx.x >> 5, lane = threadIdx.x & 31;
    int mt = blockIdx.x * 8 + wave;
    if (mt >= Mtiles) return;                 // wave-uniform, EXEC stays full
    int ntb = blockIdx.y * NB;                // first n-tile of this wave's group
    int mr = lane & 15, hg = lane >> 4;
    int m0 = mt * 16;

    v8f acc[NB];
#pragma unroll
    for (int j = 0; j < NB; ++j) acc[j] = {0.f, 0.f, 0.f, 0.f, 0.f, 0.f, 0.f, 0.f};

    const bf16_t* arow = Abase + (size_t)(m0 + mr) * aRowStride;
    for (int kt = 0; kt < KT; ++kt) {
        v16bf a = load_a_frag(arow + kt * 32 + hg * 8);
#pragma unroll
        for (int j = 0; j < NB; ++j) {
            v16bf b = *(const v16bf*)(Bp + (((size_t)(ntb + j) * KT + kt) << 9) + lane * 16);
            acc[j] = __builtin_amdgcn_wmma_f32_16x16x32_bf16(false, a, false, b,
                                                             (short)0, acc[j], false, false);
        }
    }
#pragma unroll
    for (int j = 0; j < NB; ++j) {
        int gn = (ntb + j) * 16 + mr;
        if (gn >= Nout) continue;
        float bb = (gn < Nbias) ? bias[gn] : 0.f;
#pragma unroll
        for (int r = 0; r < 8; ++r) {
            int gm = m0 + hg * 8 + r;
            float v = acc[j][r] + bb;
            if (ACT == 1) v = fmaxf(v, 0.f);
            else if (ACT == 2) v = sigf(v);
            size_t off;
            if (REMAP) {
                int s = gm % Ssz, bi = gm / Ssz;
                off = ((size_t)s * Bn + bi) * outRowStride + gn;
            } else {
                off = (size_t)gm * outRowStride + gn;
            }
            if (OUT_BF) ((bf16_t*)Out)[off] = (bf16_t)v;
            else        ((float*)Out)[off] = v;
        }
    }
}

// ---------------------------------------------------------------------------
// Highway combine: x' = gate*f_g + (1-gate)*q   (pad cols zeroed)
// ---------------------------------------------------------------------------
__global__ void k_combine(const bf16_t* __restrict__ fg, const bf16_t* __restrict__ q,
                          const bf16_t* __restrict__ gate, bf16_t* __restrict__ xout) {
    int i = blockIdx.x * 256 + threadIdx.x;
    if (i >= MROWS * DPAD) return;
    int d = i % DPAD;
    float r = 0.f;
    if (d < DIM) {
        float g = (float)gate[i];
        r = g * (float)fg[i] + (1.f - g) * (float)q[i];
    }
    xout[i] = (bf16_t)r;
}

// ---------------------------------------------------------------------------
// Persistent forward LSTM: one workgroup (32 waves) loops all 512 timesteps.
// gates_t = xg[t] + h @ Wh (WMMA, h bf16 [128][DPAD]); then c,h update.
// Tile loop stride 32 keeps mt = tile&7 wave-invariant so the compiler keeps
// all 10 h A-fragments resident in VGPRs across the whole timestep.
// Final h (t=S-1) written to hcat[:, 0:300] in f32.
// ---------------------------------------------------------------------------
__global__ void __launch_bounds__(1024)
k_lstm_fwd(const float* __restrict__ xg,        // [S][B][G]
           const bf16_t* __restrict__ Whp,      // packed, KT_D x NT_GP tiles
           bf16_t* __restrict__ h_bf,           // [B][DPAD]
           float* __restrict__ c,               // [B][DIM]
           float* __restrict__ gates,           // [B][G]
           float* __restrict__ hcat) {          // [B][2*DIM]
    const int tid = threadIdx.x;
    const int wave = tid >> 5, lane = tid & 31;
    const int mr = lane & 15, hg = lane >> 4;

    for (int i = tid; i < BATCH * DPAD; i += 1024) h_bf[i] = (bf16_t)0.f;
    for (int i = tid; i < BATCH * DIM; i += 1024) c[i] = 0.f;
    __threadfence();
    __syncthreads();

    for (int t = 0; t < SEQ; ++t) {
        const float* xgt = xg + (size_t)t * (BATCH * GDIM);
        // 8 Mtiles x 75 live Ntiles = 600 gate tiles split across 32 waves
        for (int tile = wave; tile < 8 * NT_G; tile += 32) {
            int mt = tile & 7, nt = tile >> 3;
            v8f acc = {0.f, 0.f, 0.f, 0.f, 0.f, 0.f, 0.f, 0.f};
            for (int kt = 0; kt < KT_D; ++kt) {
                const bf16_t* ap = h_bf + (size_t)(mt * 16 + mr) * DPAD + kt * 32 + hg * 8;
                v16bf a = load_a_frag(ap);
                v16bf b = *(const v16bf*)(Whp + (((size_t)nt * KT_D + kt) << 9) + lane * 16);
                acc = __builtin_amdgcn_wmma_f32_16x16x32_bf16(false, a, false, b,
                                                              (short)0, acc, false, false);
            }
            int gn = nt * 16 + mr;
#pragma unroll
            for (int r = 0; r < 8; ++r) {
                int gm = mt * 16 + hg * 8 + r;            // batch index
                size_t off = (size_t)gm * GDIM + gn;
                gates[off] = acc[r] + xgt[off];
            }
        }
        __threadfence();
        __syncthreads();
        // elementwise cell update
        for (int i = tid; i < BATCH * DIM; i += 1024) {
            int bb = i / DIM, d = i - bb * DIM;
            const float* gb = gates + (size_t)bb * GDIM;
            float ig = sigf(gb[d]);
            float fg = sigf(gb[DIM + d]);
            float gg = tanhf(gb[2 * DIM + d]);
            float og = sigf(gb[3 * DIM + d]);
            float cc = fg * c[i] + ig * gg;
            c[i] = cc;
            float hh = og * tanhf(cc);
            h_bf[(size_t)bb * DPAD + d] = (bf16_t)hh;
            if (t == SEQ - 1) hcat[(size_t)bb * (2 * DIM) + d] = hh;
        }
        __threadfence();
        __syncthreads();
    }
}

// ---------------------------------------------------------------------------
// Backward LSTM collapses to ONE step (reverse scan emits position S-1 first,
// computed from zero carry): c = sig(i)*tanh(g); h = sig(o)*tanh(c).
// ---------------------------------------------------------------------------
__global__ void k_bwd_update(const float* __restrict__ gates, float* __restrict__ hcat) {
    int i = blockIdx.x * 256 + threadIdx.x;
    if (i >= BATCH * DIM) return;
    int bb = i / DIM, d = i - bb * DIM;
    const float* gb = gates + (size_t)bb * GDIM;
    float cc = sigf(gb[d]) * tanhf(gb[2 * DIM + d]);   // f-gate * c0 == 0
    hcat[(size_t)bb * (2 * DIM) + DIM + d] = sigf(gb[3 * DIM + d]) * tanhf(cc);
}

// ---------------------------------------------------------------------------
// Head: out[b][l] = tanh(hcat[b,:600] . mean_W[:,l] + mean_b[l]) * 4
// ---------------------------------------------------------------------------
__global__ void k_final(const float* __restrict__ hcat, const float* __restrict__ Wm,
                        const float* __restrict__ bm, float* __restrict__ out) {
    int i = blockIdx.x * 256 + threadIdx.x;
    if (i >= BATCH * LATENT) return;
    int bb = i >> 6, l = i & 63;
    const float* h = hcat + (size_t)bb * (2 * DIM);
    float s = bm[l];
    for (int k = 0; k < 2 * DIM; ++k) s = fmaf(h[k], Wm[(size_t)k * LATENT + l], s);
    out[i] = tanhf(s) * 4.f;
}

// ---------------------------------------------------------------------------
extern "C" void kernel_launch(void* const* d_in, const int* in_sizes, int n_in,
                              void* d_out, int out_size, void* d_ws, size_t ws_size,
                              hipStream_t stream) {
    const int*   tokens = (const int*)d_in[0];
    const float* embed  = (const float*)d_in[1];
    const float* hw1_W  = (const float*)d_in[2];
    const float* hw1_b  = (const float*)d_in[3];
    const float* hw2_W  = (const float*)d_in[4];
    const float* hw2_b  = (const float*)d_in[5];
    const float* fwd_Wx = (const float*)d_in[6];
    const float* fwd_Wh = (const float*)d_in[7];
    const float* fwd_b  = (const float*)d_in[8];
    const float* bwd_Wx = (const float*)d_in[9];
    /* bwd_Wh (d_in[10]) unused: reverse scan's last position uses zero carry */
    const float* bwd_b  = (const float*)d_in[11];
    const float* mean_W = (const float*)d_in[12];
    const float* mean_b = (const float*)d_in[13];
    float* out = (float*)d_out;

    // ---- carve workspace ----
    size_t off = 0;
    auto alloc = [&](size_t bytes) -> char* {
        off = (off + 255) & ~(size_t)255;
        char* p = (char*)d_ws + off;
        off += bytes;
        return p;
    };
    const size_t ACT_BYTES = (size_t)MROWS * DPAD * sizeof(bf16_t);
    bf16_t* x_bf  = (bf16_t*)alloc(ACT_BYTES);
    bf16_t* x2_bf = (bf16_t*)alloc(ACT_BYTES);
    bf16_t* t_a   = (bf16_t*)alloc(ACT_BYTES);
    bf16_t* t_fg  = (bf16_t*)alloc(ACT_BYTES);
    bf16_t* t_q   = (bf16_t*)alloc(ACT_BYTES);
    const size_t WD_ELEMS = (size_t)KT_D * NT_DP * 512;  // packed 300x300 (padded N)
    const size_t WG_ELEMS = (size_t)KT_D * NT_GP * 512;  // packed 300x1200 (padded N)
    bf16_t* hw1p   = (bf16_t*)alloc(5 * WD_ELEMS * sizeof(bf16_t));
    bf16_t* hw2p   = (bf16_t*)alloc(5 * WD_ELEMS * sizeof(bf16_t));
    bf16_t* fWxp   = (bf16_t*)alloc(WG_ELEMS * sizeof(bf16_t));
    bf16_t* fWhp   = (bf16_t*)alloc(WG_ELEMS * sizeof(bf16_t));
    bf16_t* bWxp   = (bf16_t*)alloc(WG_ELEMS * sizeof(bf16_t));
    float*  xg     = (float*)alloc((size_t)SEQ * BATCH * GDIM * sizeof(float));
    float*  gatesF = (float*)alloc((size_t)BATCH * GDIM * sizeof(float));
    float*  gatesB = (float*)alloc((size_t)BATCH * GDIM * sizeof(float));
    float*  cbuf   = (float*)alloc((size_t)BATCH * DIM * sizeof(float));
    bf16_t* h_bf   = (bf16_t*)alloc((size_t)BATCH * DPAD * sizeof(bf16_t));
    float*  hcat   = (float*)alloc((size_t)BATCH * 2 * DIM * sizeof(float));

    // ---- pack weights (once per launch; deterministic) ----
    {
        dim3 gD(KT_D, NT_DP), gG(KT_D, NT_GP);
        for (int i = 0; i < 5; ++i) {
            k_pack_w<<<gD, 32, 0, stream>>>(hw1_W + (size_t)i * DIM * DIM, hw1p + i * WD_ELEMS, DIM, DIM, KT_D);
            k_pack_w<<<gD, 32, 0, stream>>>(hw2_W + (size_t)i * DIM * DIM, hw2p + i * WD_ELEMS, DIM, DIM, KT_D);
        }
        k_pack_w<<<gG, 32, 0, stream>>>(fwd_Wx, fWxp, DIM, GDIM, KT_D);
        k_pack_w<<<gG, 32, 0, stream>>>(fwd_Wh, fWhp, DIM, GDIM, KT_D);
        k_pack_w<<<gG, 32, 0, stream>>>(bwd_Wx, bWxp, DIM, GDIM, KT_D);
    }

    // ---- embedding gather -> bf16 ----
    k_embed<<<(MROWS * DPAD + 255) / 256, 256, 0, stream>>>(tokens, embed, x_bf);

    // ---- highway stages ----
    auto highway = [&](const bf16_t* xin, bf16_t* xout, const bf16_t* Wp, const float* bv) {
        dim3 g(MTILES / 8, NT_DP / NB);
        k_gemm<1, true, false><<<g, 256, 0, stream>>>(xin, DPAD, Wp + 0 * WD_ELEMS, KT_D, bv + 0 * DIM, DIM, t_a,  DPAD, DPAD, MTILES, SEQ, BATCH);
        k_gemm<1, true, false><<<g, 256, 0, stream>>>(t_a, DPAD, Wp + 1 * WD_ELEMS, KT_D, bv + 1 * DIM, DIM, t_fg, DPAD, DPAD, MTILES, SEQ, BATCH);
        k_gemm<1, true, false><<<g, 256, 0, stream>>>(xin, DPAD, Wp + 3 * WD_ELEMS, KT_D, bv + 3 * DIM, DIM, t_a,  DPAD, DPAD, MTILES, SEQ, BATCH);
        k_gemm<0, true, false><<<g, 256, 0, stream>>>(t_a, DPAD, Wp + 2 * WD_ELEMS, KT_D, bv + 2 * DIM, DIM, t_q,  DPAD, DPAD, MTILES, SEQ, BATCH);
        k_gemm<2, true, false><<<g, 256, 0, stream>>>(xin, DPAD, Wp + 4 * WD_ELEMS, KT_D, bv + 4 * DIM, DIM, t_a,  DPAD, DPAD, MTILES, SEQ, BATCH);
        k_combine<<<(MROWS * DPAD + 255) / 256, 256, 0, stream>>>(t_fg, t_q, t_a, xout);
    };
    highway(x_bf, x2_bf, hw1p, hw1_b);
    highway(x2_bf, x_bf, hw2p, hw2_b);

    // ---- xg = x @ fwd_Wx + fwd_b, written in [S][B][G] order ----
    {
        dim3 g(MTILES / 8, NT_GP / NB);
        k_gemm<0, false, true><<<g, 256, 0, stream>>>(x_bf, DPAD, fWxp, KT_D, fwd_b, GDIM,
                                                      xg, GDIM, GDIM, MTILES, SEQ, BATCH);
    }

    // ---- persistent forward recurrence ----
    k_lstm_fwd<<<1, 1024, 0, stream>>>(xg, fWhp, h_bf, cbuf, gatesF, hcat);

    // ---- backward LSTM: single step on x[:, S-1] ----
    {
        dim3 g(1, NT_GP / NB);  // Mtiles = 8 -> one 8-wave block
        k_gemm<0, false, false><<<g, 256, 0, stream>>>(x_bf + (size_t)(SEQ - 1) * DPAD,
                                                       SEQ * DPAD, bWxp, KT_D, bwd_b, GDIM,
                                                       gatesB, GDIM, GDIM, BATCH / 16, SEQ, BATCH);
    }
    k_bwd_update<<<(BATCH * DIM + 255) / 256, 256, 0, stream>>>(gatesB, hcat);

    // ---- head ----
    k_final<<<(BATCH * LATENT + 255) / 256, 256, 0, stream>>>(hcat, mean_W, mean_b, out);
}
